// CGNS_Block_5420248728082
// MI455X (gfx1250) — matrix operand order repaired and checked by reading, not compiled, not run on hardware
//
#include <hip/hip_runtime.h>
#include <hip/hip_bf16.h>
#include <math.h>

typedef _Float16 v16h __attribute__((ext_vector_type(16)));
typedef _Float16 v8h  __attribute__((ext_vector_type(8)));
typedef float    v8f  __attribute__((ext_vector_type(8)));

#define NB 4
#define NC 64
#define NN 4096
#define KK 128          // 2*C (concat of x and d^2*x rows)
#define BN_EPS 1e-5f

// ---- workspace layout (bytes); total ~149 KB ----
#define WS_A   0                          // f32[NB*NN]  a[n] = relu(tanh(w))
#define WS_D   (WS_A  + NB*NN*4)          // f32[NB*NN]  d[n] = rsqrt(a*S+1)
#define WS_S   (WS_D  + NB*NN*4)          // f32[NB*NC]  s[c] = <x[c,:], a>
#define WS_T   (WS_S  + NB*NC*4)          // f32[NB*NC]  t[c] = <x[c,:], a*d>
#define WS_AL  (WS_T  + NB*NC*4)          // f32[NB*NC]  (W0 s)*scale
#define WS_BE  (WS_AL + NB*NC*4)          // f32[NB*NC]  (W1 t)*scale
#define WS_SC  (WS_BE + NB*NC*4)          // f32[NC]     bn scale
#define WS_SH  (WS_SC + NC*4)             // f32[NC]     folded shift (conv_b,bn)
#define WS_WF  (WS_SH + NC*4)             // f16[16 frags * 32 lanes * 16] A-frags
// WS_WF == 135680, 256B aligned; wfrag size = 16384 B.

// ---------------- kernel 1: a[n], S, d[n] ----------------
__global__ __launch_bounds__(256) void k_ad(const float* __restrict__ w,
                                            float* __restrict__ a_ws,
                                            float* __restrict__ d_ws) {
  const int b = blockIdx.x, tid = threadIdx.x;
  __shared__ float red[256];
  float p = 0.f;
  for (int n = tid; n < NN; n += 256) {
    float av = fmaxf(tanhf(w[b * NN + n]), 0.f);
    a_ws[b * NN + n] = av;
    p += av;
  }
  red[tid] = p; __syncthreads();
  for (int s = 128; s > 0; s >>= 1) {
    if (tid < s) red[tid] += red[tid + s];
    __syncthreads();
  }
  const float S = red[0];
  for (int n = tid; n < NN; n += 256) {
    float av = fmaxf(tanhf(w[b * NN + n]), 0.f);   // recompute, avoid gmem RAW
    d_ws[b * NN + n] = rsqrtf(av * S + 1.0f);
  }
}

// ---------------- kernel 2: s[c], t[c] ----------------
__global__ __launch_bounds__(256) void k_st(const float* __restrict__ x,
                                            const float* __restrict__ a_ws,
                                            const float* __restrict__ d_ws,
                                            float* __restrict__ s_ws,
                                            float* __restrict__ t_ws) {
  const int b = blockIdx.x / NC, c = blockIdx.x % NC, tid = threadIdx.x;
  __shared__ float rs[256], rt[256];
  const float* xr = x + ((size_t)(b * NC + c)) * NN;
  float ps = 0.f, pt = 0.f;
  for (int n = tid; n < NN; n += 256) {
    float xv = xr[n];
    float av = a_ws[b * NN + n];
    float dv = d_ws[b * NN + n];
    ps += xv * av;
    pt += xv * av * dv;
  }
  rs[tid] = ps; rt[tid] = pt; __syncthreads();
  for (int s = 128; s > 0; s >>= 1) {
    if (tid < s) { rs[tid] += rs[tid + s]; rt[tid] += rt[tid + s]; }
    __syncthreads();
  }
  if (tid == 0) { s_ws[b * NC + c] = rs[0]; t_ws[b * NC + c] = rt[0]; }
}

// --------- kernel 3: epilogue constants + f16 A-fragment pre-pack ---------
// grid = NB blocks of 512 threads. Threads < 64 do per-channel work;
// all 512 threads of block 0 also pack conv_w into WMMA A-fragment layout.
__global__ __launch_bounds__(512) void k_prep(const float* __restrict__ conv_w,
                                              const float* __restrict__ conv_b,
                                              const float* __restrict__ bn_g,
                                              const float* __restrict__ bn_b,
                                              const float* __restrict__ bn_m,
                                              const float* __restrict__ bn_v,
                                              const float* __restrict__ s_ws,
                                              const float* __restrict__ t_ws,
                                              float* __restrict__ al_ws,
                                              float* __restrict__ be_ws,
                                              float* __restrict__ sc_ws,
                                              float* __restrict__ sh_ws,
                                              _Float16* __restrict__ wfrag) {
  const int b = blockIdx.x, tid = threadIdx.x;
  if (tid < NC) {
    const int o = tid;
    float al = 0.f, be = 0.f;
    for (int c = 0; c < NC; ++c) {
      al += conv_w[o * KK + c]      * s_ws[b * NC + c];
      be += conv_w[o * KK + NC + c] * t_ws[b * NC + c];
    }
    const float scale = bn_g[o] * rsqrtf(bn_v[o] + BN_EPS);
    al_ws[b * NC + o] = al * scale;
    be_ws[b * NC + o] = be * scale;
    if (b == 0) {
      sc_ws[o] = scale;
      sh_ws[o] = conv_b[o] * scale + bn_b[o] - bn_m[o] * scale;
    }
  }
  if (b == 0) {
    // pack A-fragments: frag(mtile 0..3, kt 0..3), lane 0..31, 16 halfs/lane
    const int lane = tid & 31, frag = tid >> 5;     // 16 frags, 512 threads
    const int mtile = frag >> 2, kt = frag & 3;
    const int m  = mtile * 16 + (lane & 15);
    const int hi = lane >> 4;
    _Float16* dst = wfrag + ((size_t)frag * 32 + lane) * 16;
#pragma unroll
    for (int j = 0; j < 16; ++j) {
      const int vg = j >> 1, slot = j & 1;
      const int k = kt * 32 + ((vg >= 4) ? 16 : 0) + hi * 8 + (vg & 3) * 2 + slot;
      dst[j] = (_Float16)conv_w[m * KK + k];
    }
  }
}

// ---------------- kernel 4: WMMA GEMM + fused rank-1 / BN / ReLU ----------
// block = 128 threads (4 waves); wave wv owns M-tile (rows wv*16..wv*16+15)
// block covers 32 output columns; grid = NB * (NN/32) = 512 blocks.
#define XPITCH 136   // halfs per LDS row (68 dwords: conflict-free b128 reads)

__global__ __launch_bounds__(128) void k_main(const float* __restrict__ x,
                                              const float* __restrict__ a_ws,
                                              const float* __restrict__ d_ws,
                                              const float* __restrict__ al_ws,
                                              const float* __restrict__ be_ws,
                                              const float* __restrict__ sc_ws,
                                              const float* __restrict__ sh_ws,
                                              const _Float16* __restrict__ wfrag,
                                              float* __restrict__ out) {
  const int blk = blockIdx.x;
  const int b   = blk >> 7;          // NN/32 = 128 column tiles per batch
  const int n0  = (blk & 127) * 32;
  const int tid = threadIdx.x;
  const int lane = tid & 31, wv = tid >> 5;

  __shared__ _Float16 xt[32 * XPITCH];   // [col][k] f16, padded pitch

  // --- stage Xcat tile: rows 0..63 = x, rows 64..127 = d^2 * x ---
  {
    const int col = tid & 31, c0 = tid >> 5;          // 4 channels per pass
    const int n = n0 + col;
    const float dv = d_ws[b * NN + n];
    const float d2 = dv * dv;
    const float* xb = x + ((size_t)b * NC) * NN;
    for (int c = c0; c < NC; c += 4) {
      const float v = xb[(size_t)c * NN + n];
      xt[col * XPITCH + c]      = (_Float16)v;
      xt[col * XPITCH + NC + c] = (_Float16)(v * d2);
    }
  }
  __syncthreads();

  // --- load pre-packed A fragments (2x global_load_b128 each) ---
  v16h afr[4];
#pragma unroll
  for (int kt = 0; kt < 4; ++kt)
    afr[kt] = *(const v16h*)(wfrag + (((size_t)wv * 4 + kt) * 32 + lane) * 16);

  const int bl = lane & 15, hi = lane >> 4;
  v8f acc[2] = {v8f{}, v8f{}};

#pragma unroll
  for (int kt = 0; kt < 4; ++kt) {
#pragma unroll
    for (int nt = 0; nt < 2; ++nt) {
      // B fragment: lane holds K = kt*32 + hi*8 .. +7 and kt*32 + 16 + hi*8 .. +7 for column nt*16+bl
      const _Float16* src = &xt[(nt * 16 + bl) * XPITCH + kt * 32 + hi * 8];
      v8h lo = *(const v8h*)(src);
      v8h hh = *(const v8h*)(src + 16);
      v16h bfr;
#pragma unroll
      for (int j = 0; j < 8; ++j) { bfr[j] = lo[j]; bfr[8 + j] = hh[j]; }
      acc[nt] = __builtin_amdgcn_wmma_f32_16x16x32_f16(
          false, afr[kt], false, bfr, (short)0, acc[nt], false, false);
    }
  }

  // --- epilogue: + a*alpha + a*d*beta, BN, ReLU ---
  const float* A2 = al_ws + b * NC;
  const float* B2 = be_ws + b * NC;
#pragma unroll
  for (int nt = 0; nt < 2; ++nt) {
    const int n = n0 + nt * 16 + bl;
    const float av = a_ws[b * NN + n];
    const float dv = d_ws[b * NN + n];
    const float ad = av * dv;
#pragma unroll
    for (int r = 0; r < 8; ++r) {
      const int o = wv * 16 + hi * 8 + r;     // D-layout: row = hi*8+r
      float y = acc[nt][r] * sc_ws[o] + av * A2[o] + ad * B2[o] + sh_ws[o];
      out[((size_t)(b * NC + o)) * NN + n] = fmaxf(y, 0.f);
    }
  }
}

extern "C" void kernel_launch(void* const* d_in, const int* in_sizes, int n_in,
                              void* d_out, int out_size, void* d_ws, size_t ws_size,
                              hipStream_t stream) {
  const float* x      = (const float*)d_in[0];
  const float* w      = (const float*)d_in[1];
  const float* conv_w = (const float*)d_in[2];
  const float* conv_b = (const float*)d_in[3];
  const float* bn_g   = (const float*)d_in[4];
  const float* bn_b   = (const float*)d_in[5];
  const float* bn_m   = (const float*)d_in[6];
  const float* bn_v   = (const float*)d_in[7];
  float* out = (float*)d_out;

  char* ws = (char*)d_ws;
  float*    a_ws  = (float*)(ws + WS_A);
  float*    d_ws_ = (float*)(ws + WS_D);
  float*    s_ws  = (float*)(ws + WS_S);
  float*    t_ws  = (float*)(ws + WS_T);
  float*    al_ws = (float*)(ws + WS_AL);
  float*    be_ws = (float*)(ws + WS_BE);
  float*    sc_ws = (float*)(ws + WS_SC);
  float*    sh_ws = (float*)(ws + WS_SH);
  _Float16* wf_ws = (_Float16*)(ws + WS_WF);

  k_ad  <<<NB,      256, 0, stream>>>(w, a_ws, d_ws_);
  k_st  <<<NB * NC, 256, 0, stream>>>(x, a_ws, d_ws_, s_ws, t_ws);
  k_prep<<<NB,      512, 0, stream>>>(conv_w, conv_b, bn_g, bn_b, bn_m, bn_v,
                                      s_ws, t_ws, al_ws, be_ws, sc_ws, sh_ws, wf_ws);
  k_main<<<NB * (NN / 32), 128, 0, stream>>>(x, a_ws, d_ws_, al_ws, be_ws,
                                             sc_ws, sh_ws, wf_ws, out);
}
